// DailyForecaster_89919435309479
// MI455X (gfx1250) — compile-verified, matrix operand order checked by reading, not run on hardware
//
#include <hip/hip_runtime.h>
#include <hip/hip_bf16.h>

typedef __bf16 bf16;
typedef __attribute__((ext_vector_type(16))) __bf16 v16bf;
typedef __attribute__((ext_vector_type(4)))  __bf16 v4bf;
typedef __attribute__((ext_vector_type(8)))  float  v8f;

#define IN_DIM   128
#define OUT_DIM  128
#define N_HEADS  4
#define HEAD_DIM 32
#define NEG_SLOPE 0.2f
#define CLAMP_MAX 20.0f
#define EPS_DEN   1e-8f

// ---- ordered-uint encoding for float atomic max ----
__device__ __forceinline__ unsigned f2key(float f) {
    unsigned b = __float_as_uint(f);
    return (b & 0x80000000u) ? ~b : (b | 0x80000000u);
}
__device__ __forceinline__ float key2f(unsigned k) {
    unsigned b = (k & 0x80000000u) ? (k & 0x7FFFFFFFu) : ~k;
    return __uint_as_float(b);
}
#define NEGINF_KEY 0x007FFFFFu  // f2key(-inf)

__device__ __forceinline__ float leaky(float v) {
    return v > 0.f ? v : NEG_SLOPE * v;
}

// ---- pack weights transposed to bf16: Wt[col][k] ----
// W: (H, IN_DIM, HEAD_DIM) fp32 ; col = head*32 + hd -> Wt[col*128 + d] = W[head][d][hd]
// W_out: (128, 128) fp32       -> Wt2[col*128 + d] = W_out[d][col]
__global__ void k_pack(const float* __restrict__ W, const float* __restrict__ W_out,
                       bf16* __restrict__ Wt, bf16* __restrict__ Wt2) {
    int idx = blockIdx.x * 256 + threadIdx.x;          // 128*128 = 16384 threads
    if (idx >= 128 * 128) return;
    int col = idx >> 7, d = idx & 127;
    int head = col >> 5, hd = col & 31;
    Wt[idx]  = (bf16)W[head * (IN_DIM * HEAD_DIM) + d * HEAD_DIM + hd];
    Wt2[idx] = (bf16)W_out[d * OUT_DIM + col];
}

// ---- projection GEMM: hb(N,128) bf16 = x(N,128)fp32 @ W(128,128), via WMMA bf16 ----
// block = 256 threads = 8 waves; wave w handles 16x16 tile at cols [16w,16w+16)
__global__ void k_proj(const float* __restrict__ x, const bf16* __restrict__ Wt,
                       bf16* __restrict__ hb, int n) {
    int rt   = blockIdx.x;
    int wave = threadIdx.x >> 5;
    int lane = threadIdx.x & 31;
    int half = lane >> 4;
    int rl   = lane & 15;
    int row  = rt * 16 + rl;
    int arow = row < n ? row : n - 1;
    const float* xr = x  + (long)arow * IN_DIM;
    const bf16*  wr = Wt + (long)(wave * 16 + rl) * IN_DIM;   // B col = rl
    int koff = half * 8;
    v8f c; 
#pragma unroll
    for (int i = 0; i < 8; ++i) c[i] = 0.f;
#pragma unroll
    for (int kk = 0; kk < IN_DIM; kk += 32) {
        v16bf a, b;
#pragma unroll
        for (int j = 0; j < 8; ++j) {
            a[j]     = (bf16)xr[kk + koff + j];
            a[8 + j] = (bf16)xr[kk + 16 + koff + j];
            b[j]     = wr[kk + koff + j];
            b[8 + j] = wr[kk + 16 + koff + j];
        }
        c = __builtin_amdgcn_wmma_f32_16x16x32_bf16(false, a, false, b,
                                                    (short)0, c, false, false);
    }
#pragma unroll
    for (int v = 0; v < 8; ++v) {
        int m = v + 8 * half;
        int orow = rt * 16 + m;
        if (orow < n) hb[(long)orow * 128 + wave * 16 + rl] = (bf16)c[v];
    }
}

// ---- per-node attention scores: s_src/s_dst (N, H) ----
__global__ void k_scores(const bf16* __restrict__ hb,
                         const float* __restrict__ a_src, const float* __restrict__ a_dst,
                         float* __restrict__ s_src, float* __restrict__ s_dst, int n) {
    int i = blockIdx.x * 256 + threadIdx.x;            // n * N_HEADS
    if (i >= n * N_HEADS) return;
    int node = i >> 2, head = i & 3;
    const bf16*  hr = hb + (long)node * 128 + head * HEAD_DIM;
    const float* as = a_src + head * HEAD_DIM;
    const float* ad = a_dst + head * HEAD_DIM;
    float ss = 0.f, sd = 0.f;
#pragma unroll
    for (int k = 0; k < HEAD_DIM; ++k) {
        float v = (float)hr[k];
        ss += v * as[k];
        sd += v * ad[k];
    }
    s_src[i] = ss;
    s_dst[i] = sd;
}

// ---- init accumulators ----
__global__ void k_init(unsigned* __restrict__ emaxk, float* __restrict__ asum,
                       float* __restrict__ agg, int n) {
    int i = blockIdx.x * 256 + threadIdx.x;
    if (i < n * 128) agg[i] = 0.f;
    if (i < n * N_HEADS) { emaxk[i] = NEGINF_KEY; asum[i] = 0.f; }
}

// ---- edge pass 1: segment max of leaky scores ----
__global__ void k_emax(const int* __restrict__ src, const int* __restrict__ dst,
                       const unsigned char* __restrict__ mask,
                       const float* __restrict__ s_src, const float* __restrict__ s_dst,
                       unsigned* __restrict__ emaxk, int E) {
    int e = blockIdx.x * 256 + threadIdx.x;
    if (e >= E) return;
    int s = src[e], d = dst[e];
    if (!mask[s]) return;
#pragma unroll
    for (int h = 0; h < N_HEADS; ++h) {
        float v = leaky(s_src[s * 4 + h] + s_dst[d * 4 + h]);
        atomicMax(&emaxk[d * 4 + h], f2key(v));
    }
}

// ---- edge pass 2: segment sum of exp ----
__global__ void k_asum(const int* __restrict__ src, const int* __restrict__ dst,
                       const unsigned char* __restrict__ mask,
                       const float* __restrict__ s_src, const float* __restrict__ s_dst,
                       const unsigned* __restrict__ emaxk, float* __restrict__ asum, int E) {
    int e = blockIdx.x * 256 + threadIdx.x;
    if (e >= E) return;
    int s = src[e], d = dst[e];
    if (!mask[s]) return;
#pragma unroll
    for (int h = 0; h < N_HEADS; ++h) {
        float v = leaky(s_src[s * 4 + h] + s_dst[d * 4 + h]);
        float num = __expf(fminf(v - key2f(emaxk[d * 4 + h]), CLAMP_MAX));
        atomicAdd(&asum[d * 4 + h], num);
    }
}

// ---- edge pass 3: weighted aggregation, one wave (32 lanes) per edge ----
__global__ void k_agg(const int* __restrict__ src, const int* __restrict__ dst,
                      const unsigned char* __restrict__ mask,
                      const float* __restrict__ s_src, const float* __restrict__ s_dst,
                      const unsigned* __restrict__ emaxk, const float* __restrict__ asum,
                      const bf16* __restrict__ hb, float* __restrict__ agg, int E) {
    long g = (long)blockIdx.x * 256 + threadIdx.x;
    int e = (int)(g >> 5);
    int lane = (int)(g & 31);
    if (e >= E) return;
    int s = src[e], d = dst[e];
    if (!mask[s]) return;
    int head = lane >> 3;                               // 8 lanes per head (4 floats each)
    float v = leaky(s_src[s * 4 + head] + s_dst[d * 4 + head]);
    float num = __expf(fminf(v - key2f(emaxk[d * 4 + head]), CLAMP_MAX));
    float alpha = num / (asum[d * 4 + head] + EPS_DEN);
    v4bf hv = *(const v4bf*)(hb + (long)s * 128 + lane * 4);
    float* ag = agg + (long)d * 128 + lane * 4;
#pragma unroll
    for (int j = 0; j < 4; ++j) atomicAdd(&ag[j], alpha * (float)hv[j]);
}

// ---- output GEMM: out(N,128)fp32 = agg(N,128)fp32 @ W_out(128,128), via WMMA bf16 ----
__global__ void k_out(const float* __restrict__ agg, const bf16* __restrict__ Wt2,
                      float* __restrict__ out, int n) {
    int rt   = blockIdx.x;
    int wave = threadIdx.x >> 5;
    int lane = threadIdx.x & 31;
    int half = lane >> 4;
    int rl   = lane & 15;
    int row  = rt * 16 + rl;
    int arow = row < n ? row : n - 1;
    const float* xr = agg + (long)arow * 128;
    const bf16*  wr = Wt2 + (long)(wave * 16 + rl) * 128;
    int koff = half * 8;
    v8f c;
#pragma unroll
    for (int i = 0; i < 8; ++i) c[i] = 0.f;
#pragma unroll
    for (int kk = 0; kk < 128; kk += 32) {
        v16bf a, b;
#pragma unroll
        for (int j = 0; j < 8; ++j) {
            a[j]     = (bf16)xr[kk + koff + j];
            a[8 + j] = (bf16)xr[kk + 16 + koff + j];
            b[j]     = wr[kk + koff + j];
            b[8 + j] = wr[kk + 16 + koff + j];
        }
        c = __builtin_amdgcn_wmma_f32_16x16x32_bf16(false, a, false, b,
                                                    (short)0, c, false, false);
    }
#pragma unroll
    for (int v = 0; v < 8; ++v) {
        int m = v + 8 * half;
        int orow = rt * 16 + m;
        if (orow < n) out[(long)orow * 128 + wave * 16 + rl] = c[v];
    }
}

extern "C" void kernel_launch(void* const* d_in, const int* in_sizes, int n_in,
                              void* d_out, int out_size, void* d_ws, size_t ws_size,
                              hipStream_t stream) {
    const float*         x     = (const float*)d_in[0];
    const int*           ei    = (const int*)d_in[1];
    const unsigned char* mask  = (const unsigned char*)d_in[2];
    const float*         W     = (const float*)d_in[3];
    const float*         a_src = (const float*)d_in[4];
    const float*         a_dst = (const float*)d_in[5];
    const float*         W_out = (const float*)d_in[6];
    float*               out   = (float*)d_out;

    const int n = in_sizes[0] / IN_DIM;   // 100000
    const int E = in_sizes[1] / 2;        // 1600000
    const int* src = ei;
    const int* dst = ei + E;

    // carve workspace (256B aligned)
    char* p = (char*)d_ws;
    auto carve = [&p](size_t bytes) -> char* {
        char* r = p;
        p += (bytes + 255) & ~(size_t)255;
        return r;
    };
    bf16*     hb    = (bf16*)carve((size_t)n * 128 * sizeof(bf16));
    bf16*     Wt    = (bf16*)carve((size_t)128 * 128 * sizeof(bf16));
    bf16*     Wt2   = (bf16*)carve((size_t)128 * 128 * sizeof(bf16));
    float*    ssrc  = (float*)carve((size_t)n * N_HEADS * sizeof(float));
    float*    sdst  = (float*)carve((size_t)n * N_HEADS * sizeof(float));
    unsigned* emaxk = (unsigned*)carve((size_t)n * N_HEADS * sizeof(unsigned));
    float*    asum  = (float*)carve((size_t)n * N_HEADS * sizeof(float));
    float*    agg   = (float*)carve((size_t)n * 128 * sizeof(float));
    (void)ws_size; (void)n_in; (void)out_size;

    const int rt = (n + 15) / 16;

    k_pack  <<<(128 * 128 + 255) / 256, 256, 0, stream>>>(W, W_out, Wt, Wt2);
    k_proj  <<<rt, 256, 0, stream>>>(x, Wt, hb, n);
    k_scores<<<(n * N_HEADS + 255) / 256, 256, 0, stream>>>(hb, a_src, a_dst, ssrc, sdst, n);
    k_init  <<<(n * 128 + 255) / 256, 256, 0, stream>>>(emaxk, asum, agg, n);
    k_emax  <<<(E + 255) / 256, 256, 0, stream>>>(src, dst, mask, ssrc, sdst, emaxk, E);
    k_asum  <<<(E + 255) / 256, 256, 0, stream>>>(src, dst, mask, ssrc, sdst, emaxk, asum, E);
    long tot = (long)E * 32;
    k_agg   <<<(int)((tot + 255) / 256), 256, 0, stream>>>(src, dst, mask, ssrc, sdst,
                                                           emaxk, asum, hb, agg, E);
    k_out   <<<rt, 256, 0, stream>>>(agg, Wt2, out, n);
}
